// SubspaceEstimator_75015898792235
// MI455X (gfx1250) — matrix operand order repaired and checked by reading, not compiled, not run on hardware
//
#include <hip/hip_runtime.h>

typedef __attribute__((ext_vector_type(16))) _Float16 v16h;
typedef __attribute__((ext_vector_type(8)))  float    v8f;

#define EMB   128
#define H1N   512
#define H2N   1024
#define H3N   512
#define NB    64
#define RANK  4
#define BATCH 64
#define SEQ   4096

// ---------------- MLP layers (tiny: 64 rows total) ----------------

__global__ __launch_bounds__(H1N) void mlp1_kernel(
    const float* __restrict__ f, const float* __restrict__ W1,
    const float* __restrict__ b1, float* __restrict__ h1) {
  __shared__ float sf[EMB];
  const int b = blockIdx.x, j = threadIdx.x;
  if (j < EMB) sf[j] = f[b * EMB + j];
  __syncthreads();
  float acc = b1[j];
#pragma unroll 8
  for (int k = 0; k < EMB; ++k) acc = fmaf(sf[k], W1[k * H1N + j], acc);
  h1[b * H1N + j] = fmaxf(acc, 0.f);
}

__global__ __launch_bounds__(H2N) void mlp2_kernel(
    const float* __restrict__ h1, const float* __restrict__ W2,
    const float* __restrict__ b2, float* __restrict__ h2) {
  __shared__ float sh[H1N];
  const int b = blockIdx.x, j = threadIdx.x;
  if (j < H1N) sh[j] = h1[b * H1N + j];
  __syncthreads();
  float acc = b2[j];
#pragma unroll 8
  for (int k = 0; k < H1N; ++k) acc = fmaf(sh[k], W2[k * H2N + j], acc);
  h2[b * H2N + j] = fmaxf(acc, 0.f);
}

__global__ __launch_bounds__(H3N) void mlp3_kernel(
    const float* __restrict__ h2, const float* __restrict__ W3,
    const float* __restrict__ b3, float* __restrict__ coeff) {
  __shared__ float sh[H2N];
  const int b = blockIdx.x, j = threadIdx.x;
  for (int i = j; i < H2N; i += H3N) sh[i] = h2[b * H2N + i];
  __syncthreads();
  float acc = b3[j];
#pragma unroll 8
  for (int k = 0; k < H2N; ++k) acc = fmaf(sh[k], W3[k * H3N + j], acc);
  coeff[b * H3N + j] = acc;  // no relu on last layer
}

// ------------- Fused basis-eval + skinny GEMM via WMMA -------------
// Per wave: one 16-seq tile. A = h_t tile (16x64 f16, generated in-reg),
// B = omega tile (64x16 f16: cols 0-3 even ranks, 4-7 odd ranks, 8-15 zero).
// C(16x16 f32) row s cols 0..7 == out[b, 2s:2s+2, 0:4] flattened -> coalesced.

__global__ __launch_bounds__(128) void basis_bmm_kernel(
    const float* __restrict__ t, const float* __restrict__ coeff,
    const float* __restrict__ mu, const float* __restrict__ alpha,
    const float* __restrict__ beta, const float* __restrict__ gamma,
    float* __restrict__ out) {
  __shared__ float s_mu[NB], s_al[NB], s_be[NB], s_ga[NB];
  __shared__ float s_cf[2 * NB * RANK];            // 512 floats: coeff row
  __shared__ __align__(16) float s_tr[4][16 * 8];  // per-wave transpose tile

  const int tid  = threadIdx.x;
  const int wave = tid >> 5;
  const int lane = tid & 31;
  const int b    = blockIdx.y;
  const int s0   = blockIdx.x * 64 + wave * 16;

  if (tid < NB) {
    s_mu[tid] = mu[tid];  s_al[tid] = alpha[tid];
    s_be[tid] = beta[tid]; s_ga[tid] = gamma[tid];
  }
  for (int i = tid; i < 2 * NB * RANK; i += 128) s_cf[i] = coeff[b * 2 * NB * RANK + i];
  __syncthreads();

  const int N  = lane & 15;   // column (B/C) or row M (A)
  const int hi = lane >> 4;   // lane half

  // ---- B fragments: K-slot h maps to K = 16*hi + h (VGPR v=h/2, half s=h%2)
  v16h bf0, bf1;
#pragma unroll
  for (int h = 0; h < 16; ++h) {
    const int k0 = hi * 16 + h;       // fragment 0: basis n = k0
    const int k1 = 32 + hi * 16 + h;  // fragment 1: basis n = 32 + k
    float v0 = 0.f, v1 = 0.f;
    if (N < RANK) {                   // even-rank block
      v0 = s_cf[k0 * RANK + N];
      v1 = s_cf[k1 * RANK + N];
    } else if (N < 2 * RANK) {        // odd-rank block
      v0 = s_cf[(NB + k0) * RANK + (N - RANK)];
      v1 = s_cf[(NB + k1) * RANK + (N - RANK)];
    }
    bf0[h] = (_Float16)v0;
    bf1[h] = (_Float16)v1;
  }

  // ---- A fragments: row M = lane&15; slot h -> K = 16*hi + h (same order as B)
  const float tv = t[(size_t)b * SEQ + s0 + N];
  v16h af0, af1;
#pragma unroll
  for (int h = 0; h < 16; ++h) {
    const int k = hi * 16 + h;  // {0..15} / {16..31}
    const int n0 = k, n1 = 32 + k;
    float d0 = s_al[n0] * (tv - s_mu[n0]);
    float d1 = s_al[n1] * (tv - s_mu[n1]);
    float v0 = __expf(-d0 * d0) * __cosf(s_be[n0] * tv + s_ga[n0]);
    float v1 = __expf(-d1 * d1) * __cosf(s_be[n1] * tv + s_ga[n1]);
    af0[h] = (_Float16)v0;
    af1[h] = (_Float16)v1;
  }

  // ---- Two chained WMMAs cover K=64
  v8f c = {};
  c = __builtin_amdgcn_wmma_f32_16x16x32_f16(false, af0, false, bf0, (short)0, c, false, false);
  c = __builtin_amdgcn_wmma_f32_16x16x32_f16(false, af1, false, bf1, (short)0, c, false, false);

  // ---- LDS transpose: lane holds col N, rows Mbase..Mbase+7 -> tile[row][col]
  if (N < 8) {
    const int Mbase = hi * 8;
#pragma unroll
    for (int r = 0; r < 8; ++r) s_tr[wave][(Mbase + r) * 8 + N] = c[r];
  }
  __syncthreads();

  // ---- Coalesced b128 stores: 32 lanes cover 16 rows x 8 cols
  const int row  = lane >> 1;
  const int half = lane & 1;
  const float4 v = *(const float4*)&s_tr[wave][row * 8 + half * 4];
  *(float4*)&out[(size_t)b * 2 * SEQ * RANK + (size_t)(s0 + row) * 2 * RANK + half * 4] = v;
}

// ------------------------------ launch ------------------------------

extern "C" void kernel_launch(void* const* d_in, const int* in_sizes, int n_in,
                              void* d_out, int out_size, void* d_ws, size_t ws_size,
                              hipStream_t stream) {
  const float* f  = (const float*)d_in[0];
  const float* t  = (const float*)d_in[1];
  const float* W1 = (const float*)d_in[2];
  const float* b1 = (const float*)d_in[3];
  const float* W2 = (const float*)d_in[4];
  const float* b2 = (const float*)d_in[5];
  const float* W3 = (const float*)d_in[6];
  const float* b3 = (const float*)d_in[7];
  const float* mu = (const float*)d_in[8];
  const float* al = (const float*)d_in[9];
  const float* be = (const float*)d_in[10];
  const float* ga = (const float*)d_in[11];
  float* out = (float*)d_out;

  float* h1    = (float*)d_ws;           // 64*512
  float* h2    = h1 + BATCH * H1N;       // 64*1024
  float* coeff = h2 + BATCH * H2N;       // 64*512

  mlp1_kernel<<<BATCH, H1N, 0, stream>>>(f, W1, b1, h1);
  mlp2_kernel<<<BATCH, H2N, 0, stream>>>(h1, W2, b2, h2);
  mlp3_kernel<<<BATCH, H3N, 0, stream>>>(h2, W3, b3, coeff);

  dim3 grid(SEQ / 64, BATCH);  // 4 waves/block, 16 seq-positions per wave
  basis_bmm_kernel<<<grid, 128, 0, stream>>>(t, coeff, mu, al, be, ga, out);
}
